// AttentionLayer_14929306321085
// MI455X (gfx1250) — matrix operand order repaired and checked by reading, not compiled, not run on hardware
//
#include <hip/hip_runtime.h>
#include <hip/hip_bf16.h>

// ---------------------------------------------------------------------------
// Attention layer for MI455X (gfx1250, wave32, WMMA).
//   qkv = x @ W^T ; q,k,v split ; A = softmax(q k^T) * 1/sqrt(64) ; out = A v
// bf16 WMMA everywhere, f32 accumulate, flash-attention fusion.
// f32->bf16 conversion is a one-shot pre-pass; GEMM loops are pure
// b128-load -> v_wmma.  Softmax row-sums are computed with a WMMA against a
// constant ones matrix.  K/V tile staging uses GLOBAL_LOAD_ASYNC_TO_LDS
// (ASYNCcnt path) when the toolchain exposes it, else vector copies.
// ---------------------------------------------------------------------------

typedef __attribute__((ext_vector_type(16))) __bf16 v16bf;
typedef __attribute__((ext_vector_type(8)))  float  v8f;
typedef __attribute__((ext_vector_type(4)))  int    v4i;   // async-LDS operand type

#define BATCH 4
#define SEQ   4096
#define DMODEL 768
#define DH    64

#if __has_builtin(__builtin_amdgcn_global_load_async_to_lds_b128) && \
    __has_builtin(__builtin_amdgcn_s_wait_asynccnt)
#define USE_ASYNC_LDS 1
#else
#define USE_ASYNC_LDS 0
#endif

static __device__ inline unsigned rne_bf16_bits(float f) {
    unsigned u = __builtin_bit_cast(unsigned, f);
    return (u + 0x7FFFu + ((u >> 16) & 1u)) >> 16;     // round-nearest-even
}

static __device__ inline __bf16 f2bf(float f) {
    return __builtin_bit_cast(__bf16, (unsigned short)rne_bf16_bits(f));
}

// Two aligned 16-byte runs -> one WMMA 16-element bf16 operand (no repacking).
static __device__ inline v16bf ld_2x16B(const __bf16* p0, const __bf16* p1) {
    struct alignas(16) U { uint4 a, b; } u;
    u.a = *(const uint4*)p0;
    u.b = *(const uint4*)p1;
    return __builtin_bit_cast(v16bf, u);
}
// One aligned 32-byte run (16 consecutive bf16).
static __device__ inline v16bf ld_32B(const __bf16* p) { return ld_2x16B(p, p + 8); }

static __device__ inline v16bf splat_ones_bf16() {
    struct alignas(16) U { uint4 a, b; } u;
    u.a = make_uint4(0x3F803F80u, 0x3F803F80u, 0x3F803F80u, 0x3F803F80u);
    u.b = u.a;
    return __builtin_bit_cast(v16bf, u);
}

static __device__ inline v8f wmma_bf16(v16bf a, v16bf b, v8f c) {
    return __builtin_amdgcn_wmma_f32_16x16x32_bf16(
        false, a, false, b, (short)0, c, false, false);
}

// ---------------------------------------------------------------------------
// Kernel 0: one-shot f32 -> bf16 conversion (bandwidth bound; 8 elts/thread).
// ---------------------------------------------------------------------------
__global__ __launch_bounds__(256)
void cvt_bf16_kernel(const float* __restrict__ src, __bf16* __restrict__ dst, long n8)
{
    const long i = (long)blockIdx.x * 256 + threadIdx.x;
    if (i >= n8) return;
    const float* p = src + i * 8;
    unsigned w[4];
    #pragma unroll
    for (int k = 0; k < 4; ++k)
        w[k] = rne_bf16_bits(p[2 * k]) | (rne_bf16_bits(p[2 * k + 1]) << 16);
    *(uint4*)(dst + i * 8) = make_uint4(w[0], w[1], w[2], w[3]);
}

// ---------------------------------------------------------------------------
// Kernel 1: QKV projection.  One wave per 16x16 tile of (B*S, 192).
// Streams A (x rows) and B (W rows) straight from L2 (both bf16 copies fit
// comfortably in the 192 MB L2); backend software-pipelines b128 loads
// against the 24-deep WMMA chain.  Outputs Q,K row-major, V transposed.
// ---------------------------------------------------------------------------
__global__ __launch_bounds__(256)
void qkv_proj_kernel(const __bf16* __restrict__ Xb, const __bf16* __restrict__ Wb,
                     __bf16* __restrict__ Q, __bf16* __restrict__ Km,
                     __bf16* __restrict__ Vt)
{
    const int lane = threadIdx.x & 31;
    const int wave = threadIdx.x >> 5;
    const int tile = blockIdx.x * 8 + wave;      // 12288 tiles = 1024 row x 12 col
    const int rowTile = tile / 12;
    const int colTile = tile % 12;
    const int r = lane & 15;
    const int g = lane >> 4;

    const __bf16* xrow = Xb + (long)(rowTile * 16 + r) * DMODEL;
    const __bf16* wrow = Wb + (long)(colTile * 16 + r) * DMODEL;

    v8f acc = (v8f){0.f,0.f,0.f,0.f,0.f,0.f,0.f,0.f};

    for (int k0 = 0; k0 < DMODEL; k0 += 32) {
        v16bf a = ld_2x16B(xrow + k0 + 8 * g, xrow + k0 + 16 + 8 * g);
        v16bf b = ld_2x16B(wrow + k0 + 8 * g, wrow + k0 + 16 + 8 * g);
        acc = wmma_bf16(a, b, acc);
    }

    // C/D layout: lane holds col n = r, rows M = j + 8g.
    const int n = colTile * 16 + r;
    #pragma unroll
    for (int j = 0; j < 8; ++j) {
        const int R  = rowTile * 16 + j + 8 * g;
        const int bi = R >> 12;
        const int s  = R & (SEQ - 1);
        const __bf16 val = f2bf(acc[j]);
        if (n < DH) {
            Q[((long)bi * SEQ + s) * DH + n] = val;
        } else if (n < 2 * DH) {
            Km[((long)bi * SEQ + s) * DH + (n - DH)] = val;
        } else {
            Vt[((long)bi * DH + (n - 2 * DH)) * SEQ + s] = val;
        }
    }
}

// ---------------------------------------------------------------------------
// Kernel 2: fused flash attention.  8 waves/WG; each wave owns 16 query rows;
// WG cooperatively stages 32-key K/V tiles in LDS (async-to-LDS when
// available).  Per chunk: 4 QK^T WMMAs, online softmax (shuffle max, WMMA
// row-sum vs ones), P restaged through LDS (C/D -> A layout), 4 PV WMMAs.
// Reference scales by 1/sqrt(DH) AFTER softmax -> final scale 0.125 / l.
// ---------------------------------------------------------------------------
__global__ __launch_bounds__(256)
void attn_kernel(const __bf16* __restrict__ Q, const __bf16* __restrict__ Km,
                 const __bf16* __restrict__ Vt, float* __restrict__ out)
{
    __shared__ __align__(16) __bf16 ldsK[32 * DH];        // 32 keys x 64 dh
    __shared__ __align__(16) __bf16 ldsV[DH * 32];        // 64 dh x 32 keys
    __shared__ __align__(16) __bf16 ldsP[8][16 * 32];     // per-wave P staging

    const int lane = threadIdx.x & 31;
    const int wave = threadIdx.x >> 5;
    const int b    = blockIdx.y;
    const int q0   = blockIdx.x * 128 + wave * 16;
    const int r = lane & 15;
    const int g = lane >> 4;

    // Q tile in WMMA A layout, two dh K-chunks (0..31, 32..63)
    const __bf16* qrow = Q + ((long)b * SEQ + q0 + r) * DH;
    v16bf qa[2];
    #pragma unroll
    for (int c = 0; c < 2; ++c)
        qa[c] = ld_2x16B(qrow + 32 * c + 8 * g, qrow + 32 * c + 16 + 8 * g);

    const v16bf kOnes = splat_ones_bf16();
    const v8f   kZero = (v8f){0.f,0.f,0.f,0.f,0.f,0.f,0.f,0.f};

    v8f o[4];
    #pragma unroll
    for (int t = 0; t < 4; ++t) o[t] = kZero;
    float m[8], l[8];
    #pragma unroll
    for (int j = 0; j < 8; ++j) { m[j] = -3.0e38f; l[j] = 0.f; }

    const __bf16* kbase = Km + (long)b * SEQ * DH;
    const __bf16* vbase = Vt + (long)b * DH * SEQ;

    for (int key0 = 0; key0 < SEQ; key0 += 32) {
        __syncthreads();   // previous chunk's LDS reads done
        {
            const int t  = threadIdx.x;
            const int dh = t >> 2, kp = (t & 3) * 8;
            const __bf16* gk = kbase + (long)key0 * DH + t * 8;   // 4KB contiguous
            const __bf16* gv = vbase + (long)dh * SEQ + key0 + kp;
#if USE_ASYNC_LDS
            __builtin_amdgcn_global_load_async_to_lds_b128(
                (v4i*)gk, (v4i*)&ldsK[t * 8], 0, 0);
            __builtin_amdgcn_global_load_async_to_lds_b128(
                (v4i*)gv, (v4i*)&ldsV[dh * 32 + kp], 0, 0);
#else
            ((uint4*)ldsK)[t] = *(const uint4*)gk;
            *(uint4*)&ldsV[dh * 32 + kp] = *(const uint4*)gv;
#endif
            if (key0 + 32 < SEQ) {   // prefetch next chunk into GL2
                __builtin_prefetch(gk + 32 * DH, 0, 1);
                __builtin_prefetch(gv + 32, 0, 1);
            }
#if USE_ASYNC_LDS
            __builtin_amdgcn_s_wait_asynccnt(0);
#endif
        }
        __syncthreads();

        // ---- QK^T: 2 x 16-key N-tiles, contraction over dh=64 ------------
        v8f s0 = kZero, s1 = kZero;
        #pragma unroll
        for (int c = 0; c < 2; ++c) {
            v16bf b0 = ld_2x16B(&ldsK[(0 * 16 + r) * DH + 32 * c + 8 * g], &ldsK[(0 * 16 + r) * DH + 32 * c + 16 + 8 * g]);
            v16bf b1 = ld_2x16B(&ldsK[(1 * 16 + r) * DH + 32 * c + 8 * g], &ldsK[(1 * 16 + r) * DH + 32 * c + 16 + 8 * g]);
            s0 = wmma_bf16(qa[c], b0, s0);
            s1 = wmma_bf16(qa[c], b1, s1);
        }

        // ---- online softmax: shuffle max, rescale, stage P ---------------
        __bf16* P = ldsP[wave];
        float corr[8];
        #pragma unroll
        for (int j = 0; j < 8; ++j) {
            float v0 = s0[j], v1 = s1[j];
            float mx = fmaxf(v0, v1);
            #pragma unroll
            for (int d = 1; d < 16; d <<= 1) mx = fmaxf(mx, __shfl_xor(mx, d, 32));
            const float mn = fmaxf(m[j], mx);
            corr[j] = __expf(m[j] - mn);
            m[j] = mn;
            const float p0 = __expf(v0 - mn);
            const float p1 = __expf(v1 - mn);
            #pragma unroll
            for (int t = 0; t < 4; ++t) o[t][j] *= corr[j];
            const int row = j + 8 * g;
            P[row * 32 + r]      = f2bf(p0);
            P[row * 32 + 16 + r] = f2bf(p1);
        }

        // ---- P in A layout; row-sums via WMMA against ones ---------------
        const __bf16* pr = &P[r * 32];
        v16bf pa = ld_2x16B(pr + 8 * g, pr + 16 + 8 * g);
        v8f rs = wmma_bf16(pa, kOnes, kZero);   // D[j][*] = rowsum(P row j+8g)
        #pragma unroll
        for (int j = 0; j < 8; ++j) l[j] = l[j] * corr[j] + rs[j];

        // ---- PV: 4 dh N-tiles --------------------------------------------
        #pragma unroll
        for (int t = 0; t < 4; ++t) {
            v16bf bv = ld_2x16B(&ldsV[(t * 16 + r) * 32 + 8 * g], &ldsV[(t * 16 + r) * 32 + 16 + 8 * g]);
            o[t] = wmma_bf16(pa, bv, o[t]);
        }
    }

    // ---- epilogue: out = (P V) * (0.125 / l)  (scale after softmax) ------
    #pragma unroll
    for (int j = 0; j < 8; ++j) {
        const float inv = 0.125f / l[j];
        const int row = q0 + j + 8 * g;
        float* orow = out + ((long)b * SEQ + row) * DH;
        #pragma unroll
        for (int t = 0; t < 4; ++t) orow[16 * t + r] = o[t][j] * inv;
    }
}

// ---------------------------------------------------------------------------
extern "C" void kernel_launch(void* const* d_in, const int* in_sizes, int n_in,
                              void* d_out, int out_size, void* d_ws, size_t ws_size,
                              hipStream_t stream)
{
    (void)in_sizes; (void)n_in; (void)out_size; (void)ws_size;
    const float* x = (const float*)d_in[0];   // (4, 4096, 768) f32
    const float* W = (const float*)d_in[1];   // (192, 768) f32
    float* out = (float*)d_out;               // (4, 4096, 64) f32

    const long NX = (long)BATCH * SEQ * DMODEL;   // 12,582,912
    const long NW = (long)3 * DH * DMODEL;        // 147,456

    __bf16* Q  = (__bf16*)d_ws;                         // 2 MB
    __bf16* Km = Q  + (size_t)BATCH * SEQ * DH;         // 2 MB
    __bf16* Vt = Km + (size_t)BATCH * SEQ * DH;         // 2 MB (transposed)
    __bf16* Xb = Vt + (size_t)BATCH * SEQ * DH;         // 24 MB (fits L2)
    __bf16* Wb = Xb + (size_t)NX;                       // 0.3 MB

    cvt_bf16_kernel<<<(int)(NX / 8 / 256), 256, 0, stream>>>(x, Xb, NX / 8);
    cvt_bf16_kernel<<<(int)(NW / 8 / 256), 256, 0, stream>>>(W, Wb, NW / 8);
    qkv_proj_kernel<<<1536, 256, 0, stream>>>(Xb, Wb, Q, Km, Vt);
    attn_kernel<<<dim3(SEQ / 128, BATCH), 256, 0, stream>>>(Q, Km, Vt, out);
}
